// GLSTM_59596966200109
// MI455X (gfx1250) — matrix operand order repaired and checked by reading, not compiled, not run on hardware
//
#include <hip/hip_runtime.h>
#include <hip/hip_bf16.h>

typedef _Float16 f16;
typedef __attribute__((ext_vector_type(16))) _Float16 v16h;
typedef __attribute__((ext_vector_type(8)))  _Float16 v8h;
typedef __attribute__((ext_vector_type(8)))  float    v8f;

#define HDIM    128
#define GDIM    512          // 4*HDIM
#define TSTEPS  12
#define NNODES  81
#define NSEQ    41472        // 512*81
#define MTILE   128          // sequences per block
#define HSTRIDE 136          // padded f16 row stride for h buffers

// ---- workspace layout (bytes) ----
#define WS_W0P   0                       // f16[ 65536]  Whh0 swizzled B-frags (128KB)
#define WS_W1P   131072                  // f16[131072]  [Wih1|Whh1] swizzled (256KB)
#define WS_XBUF  393216                  // float[NSEQ*T] (~1.9MB)

// ---- dynamic LDS layout (bytes) ----
#define SM_STG0  0                       // 32KB weight stage buf 0 (reused as FC hid)
#define SM_STG1  32768                   // 32KB weight stage buf 1
#define SM_XS    65536                   // 6KB x scalars
#define SM_B0S   71680                   // float[512] b0 = bih0+bhh0
#define SM_W0S   73728                   // float[512] wih0
#define SM_B1S   75776                   // float[512] b1 = bih1+bhh1
#define SM_H1A   77824                   // 34816B each h buffer
#define SM_H1B   112640
#define SM_H2A   147456
#define SM_H2B   182272
#define SMEM_BYTES 217088

__device__ __forceinline__ float sigmoidf_(float x) {
    return 1.0f / (1.0f + __expf(-x));
}
__device__ __forceinline__ float tanhf_(float x) {
    x = fminf(fmaxf(x, -15.0f), 15.0f);
    float e = __expf(2.0f * x);
    return (e - 1.0f) / (e + 1.0f);
}

__device__ __forceinline__ v8f wmma_f16(v16h a, v16h b, v8f c) {
    // D = A(16x32 f16) * B(32x16 f16) + C(16x16 f32)
    return __builtin_amdgcn_wmma_f32_16x16x32_f16(
        false, a, false, b, (short)0, c, false, false);
}

// CDNA5 async global->LDS copy, 16B per lane, tracked by ASYNCcnt.
__device__ __forceinline__ void async_ld16(unsigned lds_byte_off, const f16* gsrc) {
    asm volatile("global_load_async_to_lds_b128 %0, %1, off"
                 :: "v"(lds_byte_off), "v"(gsrc)
                 : "memory");
}
// ASYNCcnt completes in-order: waiting <= n (n = loads just issued for the NEXT
// chunk) guarantees the PREVIOUS chunk fully landed in LDS.
__device__ __forceinline__ void wait_async0() {
    asm volatile("s_wait_asynccnt 0x0" ::: "memory");
}
__device__ __forceinline__ void wait_async4() {
    asm volatile("s_wait_asynccnt 0x4" ::: "memory");
}
__device__ __forceinline__ void wait_async8() {
    asm volatile("s_wait_asynccnt 0x8" ::: "memory");
}

// issue one 16KB chunk (4 x 16B per thread) / 32KB chunk (8 x 16B per thread)
__device__ __forceinline__ void issue_chunk4(unsigned dst_lds, const f16* src, int tid) {
#pragma unroll
    for (int i = 0; i < 4; ++i) {
        int idx = tid + 256 * i;
        async_ld16(dst_lds + idx * 16, src + idx * 8);
    }
}
__device__ __forceinline__ void issue_chunk8(unsigned dst_lds, const f16* src, int tid) {
#pragma unroll
    for (int i = 0; i < 8; ++i) {
        int idx = tid + 256 * i;
        async_ld16(dst_lds + idx * 16, src + idx * 8);
    }
}

// A-fragment (16-bit A 16x32 layout): lane(0..15)->M, hi=lane>>4 selects K-half.
// Per lane: elements 0..7 = K[kb .. kb+7], elements 8..15 = K[kb+16 .. kb+23],
// where kb = kt*32 + 8*hi.  Two 16B LDS loads.
__device__ __forceinline__ v16h load_a_frag(const f16* hb, int row, int kb) {
    const f16* p = hb + row * HSTRIDE + kb;
    v8h lo = *(const v8h*)(p);
    v8h hi = *(const v8h*)(p + 16);
    return __builtin_shufflevector(lo, hi, 0,1,2,3,4,5,6,7,8,9,10,11,12,13,14,15);
}

// B-fragment: pre-swizzled, 512 halves per fragment (32 lanes x 16 halves), 32B/lane.
__device__ __forceinline__ v16h load_b_frag(const f16* stage, int fi, int lane) {
    return *(const v16h*)(stage + fi * 512 + lane * 16);
}

// ---------------------------------------------------------------------------
// Kernel 1: swizzle weights into WMMA B-fragment layout (f32 -> f16).
// B 32x16 layout: lane L -> N = L&15, element e -> K = 8*(L>>4) + (e<8 ? e : e+8) (+ kt*32).
// W0p chunk[jn] holds frags fi = g4*4+kt ; W1p chunk[jn] holds fi = g4*8+kt.
// ---------------------------------------------------------------------------
__global__ void prep_weights(const float* __restrict__ Whh0,
                             const float* __restrict__ Wih1,
                             const float* __restrict__ Whh1,
                             f16* __restrict__ W0p, f16* __restrict__ W1p) {
    int tid = blockIdx.x * 256 + threadIdx.x;      // 0 .. 196607
    if (tid < 65536) {                              // layer-0 Whh0 (512 x 128)
        int e    = tid & 15;
        int lane = (tid >> 4) & 31;
        int kt   = (tid >> 9) & 3;
        int g4   = (tid >> 11) & 3;
        int jn   = (tid >> 13) & 7;
        int row  = g4 * 128 + jn * 16 + (lane & 15); // gate output index j
        int k    = kt * 32 + 8 * (lane >> 4) + (e < 8 ? e : e + 8);
        W0p[tid] = (f16)Whh0[row * 128 + k];
    } else {                                        // layer-1 [Wih1|Whh1] (512 x 256)
        int t2   = tid - 65536;
        int e    = t2 & 15;
        int lane = (t2 >> 4) & 31;
        int kt   = (t2 >> 9) & 7;
        int g4   = (t2 >> 12) & 3;
        int jn   = (t2 >> 14) & 7;
        int row  = g4 * 128 + jn * 16 + (lane & 15);
        int k    = kt * 32 + 8 * (lane >> 4) + (e < 8 ? e : e + 8);
        float v  = (k < 128) ? Wih1[row * 128 + k] : Whh1[row * 128 + (k - 128)];
        W1p[t2] = (f16)v;
    }
}

// ---------------------------------------------------------------------------
// Kernel 2: income[b,n,t] = sum_a g[b,t,a] * W[a,n]  ->  xbuf[seq*T + t]
// ---------------------------------------------------------------------------
__global__ void income_kernel(const float* __restrict__ g,
                              const float* __restrict__ w,
                              float* __restrict__ xbuf) {
    int seq = blockIdx.x * 128 + threadIdx.x;       // 324*128 == NSEQ
    int b = seq / NNODES, n = seq % NNODES;
    const float* gb = g + b * TSTEPS * NNODES;
    float acc[TSTEPS];
#pragma unroll
    for (int t = 0; t < TSTEPS; ++t) acc[t] = 0.0f;
    for (int a = 0; a < NNODES; ++a) {
        float wan = w[a * NNODES + n];
#pragma unroll
        for (int t = 0; t < TSTEPS; ++t) acc[t] += gb[t * NNODES + a] * wan;
    }
#pragma unroll
    for (int t = 0; t < TSTEPS; ++t) xbuf[seq * TSTEPS + t] = acc[t];
}

// ---------------------------------------------------------------------------
// Kernel 3: fused 2-layer LSTM (T=12) + FC head.  1 block = 128 sequences,
// 8 waves, wave w owns m-tile [16w,16w+16).  c-state in VGPRs (D-layout),
// h ping-pongs through LDS f16; weights double-buffer async-staged from L2.
// Chunk order: [t: L0 jn0..7 (16KB), L1 jn0..7 (32KB)] ; chunk i -> buf[i&1].
// ---------------------------------------------------------------------------
__launch_bounds__(256, 1)
__global__ void lstm_fused(const float* __restrict__ wih0,
                           const float* __restrict__ bih0, const float* __restrict__ bhh0,
                           const float* __restrict__ bih1, const float* __restrict__ bhh1,
                           const f16* __restrict__ W0p, const f16* __restrict__ W1p,
                           const float* __restrict__ xbuf,
                           const float* __restrict__ fc1_w, const float* __restrict__ fc1_b,
                           const float* __restrict__ fc2_w, const float* __restrict__ fc2_b,
                           float* __restrict__ out) {
    extern __shared__ char smem[];
    f16*   stg0  = (f16*)(smem + SM_STG0);
    f16*   stg1  = (f16*)(smem + SM_STG1);
    float* xs    = (float*)(smem + SM_XS);
    float* b0s   = (float*)(smem + SM_B0S);
    float* w0s   = (float*)(smem + SM_W0S);
    float* b1s   = (float*)(smem + SM_B1S);
    f16*   h1b0  = (f16*)(smem + SM_H1A);
    f16*   h1b1  = (f16*)(smem + SM_H1B);
    f16*   h2b0  = (f16*)(smem + SM_H2A);
    f16*   h2b1  = (f16*)(smem + SM_H2B);
    float* hid   = (float*)(smem + SM_STG0);        // reused after time loop

    const int tid   = threadIdx.x;
    const int lane  = tid & 31;
    const int wave  = tid >> 5;
    const int l16   = lane & 15;
    const int hi    = lane >> 4;
    const int mbase = wave * 16;
    const int seq0  = blockIdx.x * MTILE;
    const unsigned stg0_lds = (unsigned)(uintptr_t)stg0;   // LDS byte offsets
    const unsigned stg1_lds = (unsigned)(uintptr_t)stg1;

    // stage x scalars for this block (128 seqs x 12 steps)
    for (int i = tid; i < MTILE * TSTEPS; i += 256)
        xs[i] = xbuf[(seq0 + i / TSTEPS) * TSTEPS + (i % TSTEPS)];
    // t-invariant per-gate constants into LDS
    for (int j = tid; j < GDIM; j += 256) {
        b0s[j] = bih0[j] + bhh0[j];
        b1s[j] = bih1[j] + bhh1[j];
        w0s[j] = wih0[j];
    }
    // zero h(-1) buffers
    for (int i = tid; i < MTILE * HSTRIDE; i += 256) {
        h1b0[i] = (f16)0.0f;
        h2b0[i] = (f16)0.0f;
    }

    // cell states, D-layout: c[jn][r] for column jn*16+l16, row mbase + r + 8*hi
    v8f c1[8], c2[8];
#pragma unroll
    for (int jn = 0; jn < 8; ++jn) { c1[jn] = (v8f)(0.0f); c2[jn] = (v8f)(0.0f); }

    // prologue prefetch: L0 chunk jn=0 -> buf0
    issue_chunk4(stg0_lds, W0p, tid);

    __syncthreads();

    for (int t = 0; t < TSTEPS; ++t) {
        const f16* h1p = (t & 1) ? h1b1 : h1b0;
        f16*       h1c = (t & 1) ? h1b0 : h1b1;
        const f16* h2p = (t & 1) ? h2b1 : h2b0;
        f16*       h2c = (t & 1) ? h2b0 : h2b1;

        // ================= LAYER 0 : gates = x*wih0 + h1p @ Whh0^T + b0 =====
#pragma unroll
        for (int jn = 0; jn < 8; ++jn) {
            // prefetch next chunk into the other buffer, then wait for current
            if (jn < 7) {
                issue_chunk4((jn & 1) ? stg0_lds : stg1_lds, W0p + (jn + 1) * 8192, tid);
                wait_async4();
            } else {                               // next = L1 jn0 (32KB) -> buf0
                issue_chunk8(stg0_lds, W1p, tid);
                wait_async8();
            }
            __syncthreads();
            const f16* stage = (jn & 1) ? stg1 : stg0;

            float bb[4], ww[4];
#pragma unroll
            for (int g4 = 0; g4 < 4; ++g4) {
                bb[g4] = b0s[g4 * 128 + jn * 16 + l16];
                ww[g4] = w0s[g4 * 128 + jn * 16 + l16];
            }
            float xv[8];
#pragma unroll
            for (int r = 0; r < 8; ++r)
                xv[r] = xs[(mbase + r + 8 * hi) * TSTEPS + t];

            v8f a0, a1, a2, a3;                     // i,f,g,o accumulators
#pragma unroll
            for (int r = 0; r < 8; ++r) {
                a0[r] = bb[0] + xv[r] * ww[0];
                a1[r] = bb[1] + xv[r] * ww[1];
                a2[r] = bb[2] + xv[r] * ww[2];
                a3[r] = bb[3] + xv[r] * ww[3];
            }
#pragma unroll
            for (int kt = 0; kt < 4; ++kt) {
                v16h b0 = load_b_frag(stage, 0 * 4 + kt, lane);
                v16h b1 = load_b_frag(stage, 1 * 4 + kt, lane);
                v16h b2 = load_b_frag(stage, 2 * 4 + kt, lane);
                v16h b3 = load_b_frag(stage, 3 * 4 + kt, lane);
                v16h a  = load_a_frag(h1p, mbase + l16, kt * 32 + 8 * hi);
                a0 = wmma_f16(a, b0, a0);
                a1 = wmma_f16(a, b1, a1);
                a2 = wmma_f16(a, b2, a2);
                a3 = wmma_f16(a, b3, a3);
            }
#pragma unroll
            for (int r = 0; r < 8; ++r) {
                float ig = sigmoidf_(a0[r]);
                float fg = sigmoidf_(a1[r]);
                float gg = tanhf_(a2[r]);
                float og = sigmoidf_(a3[r]);
                float c  = fg * c1[jn][r] + ig * gg;
                c1[jn][r] = c;
                float h  = og * tanhf_(c);
                h1c[(mbase + r + 8 * hi) * HSTRIDE + jn * 16 + l16] = (f16)h;
            }
            __syncthreads();                        // done reading stage buf[jn&1]
        }

        // ======= LAYER 1 : gates = [h1c | h2p] @ [Wih1|Whh1]^T + b1 (K=256) ==
#pragma unroll
        for (int jn = 0; jn < 8; ++jn) {
            if (jn < 7) {
                issue_chunk8((jn & 1) ? stg0_lds : stg1_lds, W1p + (jn + 1) * 16384, tid);
                wait_async8();
            } else {                               // next = next-t L0 jn0 -> buf0
                issue_chunk4(stg0_lds, W0p, tid);   // (redundant at t==11; drained below)
                wait_async4();
            }
            __syncthreads();                        // h1c also complete at jn==0
            const f16* stage = (jn & 1) ? stg1 : stg0;

            float bb[4];
#pragma unroll
            for (int g4 = 0; g4 < 4; ++g4)
                bb[g4] = b1s[g4 * 128 + jn * 16 + l16];

            v8f a0, a1, a2, a3;
#pragma unroll
            for (int r = 0; r < 8; ++r) {
                a0[r] = bb[0]; a1[r] = bb[1];
                a2[r] = bb[2]; a3[r] = bb[3];
            }
#pragma unroll
            for (int kt = 0; kt < 8; ++kt) {
                const f16* hsrc = (kt < 4) ? h1c : h2p;
                int kb = (kt & 3) * 32 + 8 * hi;
                v16h b0 = load_b_frag(stage, 0 * 8 + kt, lane);
                v16h b1 = load_b_frag(stage, 1 * 8 + kt, lane);
                v16h b2 = load_b_frag(stage, 2 * 8 + kt, lane);
                v16h b3 = load_b_frag(stage, 3 * 8 + kt, lane);
                v16h a  = load_a_frag(hsrc, mbase + l16, kb);
                a0 = wmma_f16(a, b0, a0);
                a1 = wmma_f16(a, b1, a1);
                a2 = wmma_f16(a, b2, a2);
                a3 = wmma_f16(a, b3, a3);
            }
#pragma unroll
            for (int r = 0; r < 8; ++r) {
                float ig = sigmoidf_(a0[r]);
                float fg = sigmoidf_(a1[r]);
                float gg = tanhf_(a2[r]);
                float og = sigmoidf_(a3[r]);
                float c  = fg * c2[jn][r] + ig * gg;
                c2[jn][r] = c;
                float h  = og * tanhf_(c);
                h2c[(mbase + r + 8 * hi) * HSTRIDE + jn * 16 + l16] = (f16)h;
            }
            __syncthreads();                        // done reading stage buf[jn&1]
        }
    }

    // drain the tail prefetch before reusing stg0 as the FC hid buffer
    wait_async0();
    __syncthreads();

    // ======================= FC head on h2[T-1] ============================
    const f16* hf = h2b0;                           // T=12 even -> last write in buf0
    {
        int m  = tid >> 1;
        int qh = (tid & 1) * 16;
#pragma unroll
        for (int q0 = 0; q0 < 16; ++q0) {
            int q = qh + q0;
            float s = fc1_b[q];
            for (int j = 0; j < HDIM; ++j)
                s += (float)hf[m * HSTRIDE + j] * fc1_w[q * HDIM + j];
            hid[m * 32 + q] = fmaxf(s, 0.0f);
        }
    }
    __syncthreads();
    if (tid < MTILE) {
        float s = fc2_b[0];
#pragma unroll
        for (int q = 0; q < 32; ++q) s += hid[tid * 32 + q] * fc2_w[q];
        out[seq0 + tid] = fmaxf(s, 0.0f);
    }
}

// ---------------------------------------------------------------------------
extern "C" void kernel_launch(void* const* d_in, const int* in_sizes, int n_in,
                              void* d_out, int out_size, void* d_ws, size_t ws_size,
                              hipStream_t stream) {
    const float* g      = (const float*)d_in[0];
    const float* wgt    = (const float*)d_in[1];
    const float* Wih0   = (const float*)d_in[2];
    const float* Whh0   = (const float*)d_in[3];
    const float* bih0   = (const float*)d_in[4];
    const float* bhh0   = (const float*)d_in[5];
    const float* Wih1   = (const float*)d_in[6];
    const float* Whh1   = (const float*)d_in[7];
    const float* bih1   = (const float*)d_in[8];
    const float* bhh1   = (const float*)d_in[9];
    const float* fc1_w  = (const float*)d_in[10];
    const float* fc1_b  = (const float*)d_in[11];
    const float* fc2_w  = (const float*)d_in[12];
    const float* fc2_b  = (const float*)d_in[13];

    char*  ws   = (char*)d_ws;
    f16*   W0p  = (f16*)(ws + WS_W0P);
    f16*   W1p  = (f16*)(ws + WS_W1P);
    float* xbuf = (float*)(ws + WS_XBUF);

    prep_weights<<<768, 256, 0, stream>>>(Whh0, Wih1, Whh1, W0p, W1p);
    income_kernel<<<NSEQ / 128, 128, 0, stream>>>(g, wgt, xbuf);
    lstm_fused<<<NSEQ / MTILE, 256, SMEM_BYTES, stream>>>(
        Wih0, bih0, bhh0, bih1, bhh1, W0p, W1p, xbuf,
        fc1_w, fc1_b, fc2_w, fc2_b, (float*)d_out);
}
